// TransformerBlock_24103356465245
// MI455X (gfx1250) — matrix operand order repaired and checked by reading, not compiled, not run on hardware
//
#include <hip/hip_runtime.h>

// ---------------------------------------------------------------------------
// Types for CDNA5 WMMA (wave32, 16x16x32 bf16 -> f32)
// ---------------------------------------------------------------------------
typedef __bf16 bf16;
typedef bf16         v16bf __attribute__((ext_vector_type(16)));
typedef float        v8f   __attribute__((ext_vector_type(8)));
typedef unsigned int v4u   __attribute__((ext_vector_type(4)));

union Frag {          // 32 bytes: one WMMA 16-bit operand = two b128 chunks
    v16bf v;
    v4u   q[2];
};

static __device__ __forceinline__ v8f wmma_bf16(v16bf a, v16bf b, v8f c) {
    return __builtin_amdgcn_wmma_f32_16x16x32_bf16(false, a, false, b, (short)0, c,
                                                   false, false);
}
static __device__ __forceinline__ v4u ld128(const bf16* p) {
    return *(const v4u*)(p);
}

// Model constants (match reference setup_inputs)
#define TT 2048
#define DD 1024
#define EE 8
#define HH 2730
#define HP 2752          /* H padded to 43*64 so GEMMs are tail-free */
#define NHEAD 16
#define HDIM 64

// ---------------------------------------------------------------------------
// fp32 -> bf16 cast (grid-stride)
// ---------------------------------------------------------------------------
__global__ void cast_f32_bf16(const float* __restrict__ in, bf16* __restrict__ out,
                              size_t n) {
    size_t i = (size_t)blockIdx.x * blockDim.x + threadIdx.x;
    size_t stride = (size_t)gridDim.x * blockDim.x;
    for (; i < n; i += stride) out[i] = (bf16)in[i];
}

// ---------------------------------------------------------------------------
// Transpose + zero-pad + cast: B[K,N] f32 (row-major) -> Bt[Np,Kp] bf16 where
// Bt[n*Kp + k] = (n<N && k<K) ? B[k*N + n] : 0.   Grid: (ceil(Kp/256), Np)
// ---------------------------------------------------------------------------
__global__ __launch_bounds__(256) void transpose_pad_cast(
    const float* __restrict__ in, bf16* __restrict__ out, int K, int N, int Kp) {
    int n = blockIdx.y;
    int k = blockIdx.x * 256 + threadIdx.x;
    if (k >= Kp) return;
    float v = (n < N && k < K) ? in[(size_t)k * N + n] : 0.0f;
    out[(size_t)n * Kp + k] = (bf16)v;
}

// ---------------------------------------------------------------------------
// LayerNorm row kernel: out_bf16[row] = (x-mu)*rsqrt(var+eps)*w
// ---------------------------------------------------------------------------
__global__ __launch_bounds__(256) void layernorm_bf16(const float* __restrict__ x,
                                                      const float* __restrict__ w,
                                                      bf16* __restrict__ out, int D) {
    int row = blockIdx.x;
    const float* xr = x + (size_t)row * D;
    float s = 0.f, s2 = 0.f;
    for (int i = threadIdx.x; i < D; i += 256) {
        float v = xr[i];
        s += v; s2 += v * v;
    }
    __shared__ float r1[256], r2[256];
    r1[threadIdx.x] = s; r2[threadIdx.x] = s2;
    __syncthreads();
    for (int off = 128; off > 0; off >>= 1) {
        if (threadIdx.x < off) {
            r1[threadIdx.x] += r1[threadIdx.x + off];
            r2[threadIdx.x] += r2[threadIdx.x + off];
        }
        __syncthreads();
    }
    float mu = r1[0] / (float)D;
    float var = r2[0] / (float)D - mu * mu;
    float rs = rsqrtf(var + 1e-5f);
    for (int i = threadIdx.x; i < D; i += 256)
        out[(size_t)row * D + i] = (bf16)((xr[i] - mu) * rs * w[i]);
}

// ---------------------------------------------------------------------------
// WMMA GEMM: C[M,N](f32) = A[M,K](bf16 row-major) * Bt[N,K](bf16, B col-major)
// Requirements: M%128==0, N%64==0, K%32==0 (guaranteed by padding).
// Block = 256 threads = 8 waves; block tile 128(M) x 64(N); wave = 16x64 strip.
// Software pipeline: next k-step's global loads (B stage + A frags) are issued
// before the current step's WMMAs; LDS B tile is double-buffered so only ONE
// barrier per k-step and the ds_store of tile k+1 lands after the math,
// hiding global latency behind 4 WMMAs. All fragments are b128 transfers.
// ---------------------------------------------------------------------------
__global__ __launch_bounds__(256) void gemm_bf16_wmma(
    const bf16* __restrict__ A, const bf16* __restrict__ Bt, float* __restrict__ C,
    int M, int N, int K,
    const float* __restrict__ addSrc,       // nullable: + addSrc[row*N+col]
    const float* __restrict__ rowScale,     // nullable: * rowScale[row*rsStride]
    int rsStride, int accumulate) {
    __shared__ __attribute__((aligned(16))) bf16 Bl[2][64 * 32];  // double buffer

    int tid  = threadIdx.x;
    int wid  = tid >> 5;
    int lane = tid & 31;
    int hi = lane >> 4, ln = lane & 15;
    int mBase = blockIdx.y * 128 + wid * 16;
    int nBase = blockIdx.x * 64;

    const bf16* Ar = A + (size_t)(mBase + ln) * K;
    int kb = hi * 8;                       // A per-lane-half K base (ISA 7.12.2)

    // staging map: thread t loads 8 bf16 of column (nBase + t/4), k-chunk t%4
    int sCol = tid >> 2;
    int sOff = (tid & 3) * 8;
    int sIdx = sCol * 32 + sOff;
    const bf16* Bs = Bt + (size_t)(nBase + sCol) * K + sOff;

    v8f acc[4];
#pragma unroll
    for (int nt = 0; nt < 4; ++nt)
#pragma unroll
        for (int r = 0; r < 8; ++r) acc[nt][r] = 0.f;

    int kTiles = K >> 5;

    // ---- pipeline prologue: tile 0 ----
    v4u bqN = ld128(Bs);
    Frag aN;
    aN.q[0] = ld128(Ar + kb);
    aN.q[1] = ld128(Ar + kb + 16);
    *(v4u*)(&Bl[0][sIdx]) = bqN;
    __syncthreads();

    for (int kt = 0; kt < kTiles; ++kt) {
        int cur = kt & 1;
        Frag a = aN;
        // issue next tile's global loads early (latency hidden behind WMMAs)
        int k1 = (kt + 1 < kTiles) ? (kt + 1) * 32 : 0;
        bqN = ld128(Bs + k1);
        __builtin_prefetch(Bs + k1 + 32, 0, 1);       // global_prefetch_b8
        aN.q[0] = ld128(Ar + k1 + kb);
        aN.q[1] = ld128(Ar + k1 + kb + 16);

        // batch all 8 ds_load_b128, then 4 back-to-back WMMAs
        Frag b[4];
#pragma unroll
        for (int nt = 0; nt < 4; ++nt) {
            const bf16* bp = &Bl[cur][(nt * 16 + ln) * 32 + hi * 8];
            b[nt].q[0] = *(const v4u*)(bp);
            b[nt].q[1] = *(const v4u*)(bp + 16);
        }
#pragma unroll
        for (int nt = 0; nt < 4; ++nt)
            acc[nt] = wmma_bf16(a.v, b[nt].v, acc[nt]);

        // stage tile k+1 into the other buffer; single barrier per k-step
        *(v4u*)(&Bl[1 - cur][sIdx]) = bqN;
        __syncthreads();
    }

#pragma unroll
    for (int nt = 0; nt < 4; ++nt) {
        int col = nBase + nt * 16 + ln;
#pragma unroll
        for (int r = 0; r < 8; ++r) {
            int row = mBase + r + hi * 8;
            size_t idx = (size_t)row * N + col;
            float v = acc[nt][r];
            if (rowScale) v *= rowScale[(size_t)row * rsStride];
            if (addSrc)   v += addSrc[idx];
            if (accumulate) C[idx] += v; else C[idx] = v;
        }
    }
}

// ---------------------------------------------------------------------------
// RoPE: qf/kf [T,D] fp32 -> qb/kb [T,D] bf16, rotate-half convention
// ---------------------------------------------------------------------------
__global__ void rope_bf16(const float* __restrict__ qf, const float* __restrict__ kf,
                          bf16* __restrict__ qb, bf16* __restrict__ kb) {
    int idx = blockIdx.x * blockDim.x + threadIdx.x;
    if (idx >= TT * NHEAD * 32) return;
    int i = idx & 31;
    int h = (idx >> 5) & (NHEAD - 1);
    int t = idx >> 9;
    float ex = (2.0f * (float)i) / (float)HDIM;
    float ang = (float)t * __expf(-ex * 9.210340372f);   // 10000^-ex
    float c = __cosf(ang), s = __sinf(ang);
    size_t base = (size_t)t * DD + h * HDIM;
    float q1 = qf[base + i], q2 = qf[base + 32 + i];
    float k1 = kf[base + i], k2 = kf[base + 32 + i];
    qb[base + i]      = (bf16)(q1 * c - q2 * s);
    qb[base + 32 + i] = (bf16)(q2 * c + q1 * s);
    kb[base + i]      = (bf16)(k1 * c - k2 * s);
    kb[base + 32 + i] = (bf16)(k2 * c + k1 * s);
}

// ---------------------------------------------------------------------------
// Flash attention, causal. One 32-thread (1-wave) block per (head, 16-query
// tile). QK^T and PV via WMMA; online softmax. K is naturally "column-major
// B" for QK^T (hd contiguous per key); V is pre-transposed per head
// (Vt[d][t]) so every fragment is b128 loads. P goes C-layout -> A-layout
// through 1KB LDS. T%32==0 => no tail masking needed on loads.
// ---------------------------------------------------------------------------
__global__ __launch_bounds__(32) void flash_attn_wmma(
    const bf16* __restrict__ Qb, const bf16* __restrict__ Kb,
    const bf16* __restrict__ Vt,   // [D, T]: Vt[(h*64+n)*T + t]
    float* __restrict__ O) {
    const int numQT = TT / 16;
    int head = blockIdx.x / numQT;
    int qt   = blockIdx.x % numQT;
    int lane = threadIdx.x;
    int hi = lane >> 4, ln = lane & 15;
    int qrow0 = qt * 16;
    int kb = hi * 8;
    size_t hoff = (size_t)head * HDIM;

    // Q A-fragments for the two hd k-steps, loaded once (4 x b128)
    Frag qa[2];
    {
        const bf16* Qr = Qb + (size_t)(qrow0 + ln) * DD + hoff;
#pragma unroll
        for (int s = 0; s < 2; ++s) {
            qa[s].q[0] = ld128(Qr + s * 32 + kb);
            qa[s].q[1] = ld128(Qr + s * 32 + kb + 16);
        }
    }

    v8f oacc[4];
    float mrow[8], lrow[8];
#pragma unroll
    for (int nt = 0; nt < 4; ++nt)
#pragma unroll
        for (int r = 0; r < 8; ++r) oacc[nt][r] = 0.f;
#pragma unroll
    for (int r = 0; r < 8; ++r) { mrow[r] = -3.0e38f; lrow[r] = 0.f; }

    __shared__ __attribute__((aligned(16))) bf16 Pl[16 * 32];

    int nKB = (qrow0 + 16 + 31) / 32;      // causal: keys < qrow0+16
    for (int kbIdx = 0; kbIdx < nKB; ++kbIdx) {
        int key0 = kbIdx * 32;
        // ---- scores: two 16-key subtiles ----
        float sc[2][8];
#pragma unroll
        for (int st = 0; st < 2; ++st) {
            v8f acc = {0.f, 0.f, 0.f, 0.f, 0.f, 0.f, 0.f, 0.f};
            const bf16* Kr = Kb + (size_t)(key0 + st * 16 + ln) * DD + hoff;
#pragma unroll
            for (int s = 0; s < 2; ++s) {
                Frag bk;
                bk.q[0] = ld128(Kr + s * 32 + hi * 8);
                bk.q[1] = ld128(Kr + s * 32 + hi * 8 + 16);
                acc = wmma_bf16(qa[s].v, bk.v, acc);
            }
#pragma unroll
            for (int r = 0; r < 8; ++r) sc[st][r] = acc[r];
        }
        // ---- online softmax + write P to LDS ----
#pragma unroll
        for (int r = 0; r < 8; ++r) {
            int qr = qrow0 + r + hi * 8;
            float v0 = sc[0][r] * 0.125f;           // 1/sqrt(64)
            float v1 = sc[1][r] * 0.125f;
            v0 = (key0 + ln      > qr) ? -1.0e30f : v0;   // value-select only
            v1 = (key0 + 16 + ln > qr) ? -1.0e30f : v1;
            float m = fmaxf(v0, v1);
            for (int off = 8; off > 0; off >>= 1)
                m = fmaxf(m, __shfl_xor(m, off, 32));
            float nm = fmaxf(mrow[r], m);
            float corr = __expf(mrow[r] - nm);
            float p0 = __expf(v0 - nm);
            float p1 = __expf(v1 - nm);
            float ls = p0 + p1;
            for (int off = 8; off > 0; off >>= 1)
                ls += __shfl_xor(ls, off, 32);
            lrow[r] = lrow[r] * corr + ls;
            mrow[r] = nm;
#pragma unroll
            for (int nt = 0; nt < 4; ++nt) oacc[nt][r] *= corr;
            int prow = r + hi * 8;
            Pl[prow * 32 + ln]      = (bf16)p0;
            Pl[prow * 32 + 16 + ln] = (bf16)p1;
        }
        __syncthreads();
        // ---- PV: P(16x32) x V(32x64) ----
        Frag pa;
        pa.q[0] = *(const v4u*)(&Pl[ln * 32 + kb]);
        pa.q[1] = *(const v4u*)(&Pl[ln * 32 + kb + 16]);
#pragma unroll
        for (int nt = 0; nt < 4; ++nt) {
            Frag bv;
            const bf16* Vr = Vt + (hoff + nt * 16 + ln) * TT + key0;
            bv.q[0] = ld128(Vr + hi * 8);
            bv.q[1] = ld128(Vr + hi * 8 + 16);
            oacc[nt] = wmma_bf16(pa.v, bv.v, oacc[nt]);
        }
        __syncthreads();
    }
    // ---- epilogue ----
#pragma unroll
    for (int r = 0; r < 8; ++r) {
        float inv = 1.0f / lrow[r];
        int row = qrow0 + r + hi * 8;
#pragma unroll
        for (int nt = 0; nt < 4; ++nt)
            O[(size_t)row * DD + hoff + nt * 16 + ln] = oacc[nt][r] * inv;
    }
}

// ---------------------------------------------------------------------------
// Router: logits = h2 @ gate_w [N,8], softmax, top-2, renormalize -> combine
// ---------------------------------------------------------------------------
__global__ __launch_bounds__(32) void gate_topk(const bf16* __restrict__ h2,
                                                const float* __restrict__ gw,
                                                float* __restrict__ comb) {
    int n = blockIdx.x;
    int lane = threadIdx.x;
    float acc[EE];
#pragma unroll
    for (int e = 0; e < EE; ++e) acc[e] = 0.f;
    for (int d = lane; d < DD; d += 32) {
        float h = (float)h2[(size_t)n * DD + d];
#pragma unroll
        for (int e = 0; e < EE; ++e) acc[e] += h * gw[(size_t)d * EE + e];
    }
#pragma unroll
    for (int e = 0; e < EE; ++e)
        for (int off = 16; off > 0; off >>= 1)
            acc[e] += __shfl_xor(acc[e], off, 32);
    if (lane == 0) {
        float m = acc[0];
        for (int e = 1; e < EE; ++e) m = fmaxf(m, acc[e]);
        float p[EE], sum = 0.f;
        for (int e = 0; e < EE; ++e) { p[e] = __expf(acc[e] - m); sum += p[e]; }
        float inv = 1.0f / sum;
        for (int e = 0; e < EE; ++e) p[e] *= inv;
        int i0 = 0;
        for (int e = 1; e < EE; ++e) if (p[e] > p[i0]) i0 = e;
        int i1 = (i0 == 0) ? 1 : 0;
        for (int e = 0; e < EE; ++e) if (e != i0 && p[e] > p[i1]) i1 = e;
        float wsum = p[i0] + p[i1];
        for (int e = 0; e < EE; ++e) {
            float w = 0.f;
            if (e == i0) w = p[i0] / wsum;
            else if (e == i1) w = p[i1] / wsum;
            comb[(size_t)n * EE + e] = w;
        }
    }
}

// ---------------------------------------------------------------------------
// SwiGLU fuse: act = silu(c1) * c3  -> bf16 (zeros propagate through padding)
// ---------------------------------------------------------------------------
__global__ void silu_mul_bf16(const float* __restrict__ c1, const float* __restrict__ c3,
                              bf16* __restrict__ out, size_t n) {
    size_t i = (size_t)blockIdx.x * blockDim.x + threadIdx.x;
    size_t stride = (size_t)gridDim.x * blockDim.x;
    for (; i < n; i += stride) {
        float g = c1[i];
        out[i] = (bf16)((g / (1.0f + __expf(-g))) * c3[i]);
    }
}

// ---------------------------------------------------------------------------
// Host-side orchestration
// ---------------------------------------------------------------------------
static inline int ceilDiv(int a, int b) { return (a + b - 1) / b; }

extern "C" void kernel_launch(void* const* d_in, const int* in_sizes, int n_in,
                              void* d_out, int out_size, void* d_ws, size_t ws_size,
                              hipStream_t stream) {
    const float* x     = (const float*)d_in[0];
    const float* ln1_w = (const float*)d_in[1];
    const float* ln2_w = (const float*)d_in[2];
    const float* wq    = (const float*)d_in[3];
    const float* wk    = (const float*)d_in[4];
    const float* wv    = (const float*)d_in[5];
    const float* wo    = (const float*)d_in[6];
    const float* gatew = (const float*)d_in[7];
    const float* w1    = (const float*)d_in[8];
    const float* w2    = (const float*)d_in[9];
    const float* w3    = (const float*)d_in[10];
    float* out = (float*)d_out;
    (void)ws_size; (void)in_sizes; (void)n_in; (void)out_size;

    // workspace bump allocator
    char* p = (char*)d_ws;
    auto alloc = [&](size_t bytes) -> void* {
        void* r = (void*)p;
        p += (bytes + 255) & ~(size_t)255;
        return r;
    };
    const size_t TD = (size_t)TT * DD;
    const size_t THp = (size_t)TT * HP;

    bf16* wqT = (bf16*)alloc((size_t)DD * DD * 2);   // [N=1024, K=1024]
    bf16* wkT = (bf16*)alloc((size_t)DD * DD * 2);
    bf16* wvT = (bf16*)alloc((size_t)DD * DD * 2);
    bf16* woT = (bf16*)alloc((size_t)DD * DD * 2);
    bf16* w1T = (bf16*)alloc((size_t)EE * HP * DD * 2);   // per-e: [HP, 1024]
    bf16* w3T = (bf16*)alloc((size_t)EE * HP * DD * 2);
    bf16* w2T = (bf16*)alloc((size_t)EE * DD * HP * 2);   // per-e: [1024, HP]
    bf16* h1B = (bf16*)alloc(TD * 2);
    float* qF = (float*)alloc(TD * 4);
    float* kF = (float*)alloc(TD * 4);
    float* vF = (float*)alloc(TD * 4);
    bf16* qB = (bf16*)alloc(TD * 2);
    bf16* kB = (bf16*)alloc(TD * 2);
    bf16* vT = (bf16*)alloc(TD * 2);                 // [D, T]
    float* oF = (float*)alloc(TD * 4);
    bf16* oB = (bf16*)alloc(TD * 2);
    bf16* h2B = (bf16*)alloc(TD * 2);
    float* comb = (float*)alloc((size_t)TT * EE * 4);
    float* c1 = (float*)alloc(THp * 4);
    float* c3 = (float*)alloc(THp * 4);
    bf16* actB = (bf16*)alloc(THp * 2);

    // --- one-time weight transpose-casts (bf16 results live in 192MB L2) ---
    dim3 gT1(ceilDiv(DD, 256), DD);          // [1024,1024] -> [1024,1024]
    transpose_pad_cast<<<gT1, 256, 0, stream>>>(wq, wqT, DD, DD, DD);
    transpose_pad_cast<<<gT1, 256, 0, stream>>>(wk, wkT, DD, DD, DD);
    transpose_pad_cast<<<gT1, 256, 0, stream>>>(wv, wvT, DD, DD, DD);
    transpose_pad_cast<<<gT1, 256, 0, stream>>>(wo, woT, DD, DD, DD);
    for (int e = 0; e < EE; ++e) {
        // w1,w3: B=[K=D,N=H] -> Bt[HP, D]
        dim3 gUpT(ceilDiv(DD, 256), HP);
        transpose_pad_cast<<<gUpT, 256, 0, stream>>>(w1 + (size_t)e * DD * HH,
                                                     w1T + (size_t)e * HP * DD, DD, HH, DD);
        transpose_pad_cast<<<gUpT, 256, 0, stream>>>(w3 + (size_t)e * DD * HH,
                                                     w3T + (size_t)e * HP * DD, DD, HH, DD);
        // w2: B=[K=H,N=D] -> Bt[D, HP] (zero-pad K)
        dim3 gDnT(ceilDiv(HP, 256), DD);
        transpose_pad_cast<<<gDnT, 256, 0, stream>>>(w2 + (size_t)e * HH * DD,
                                                     w2T + (size_t)e * DD * HP, HH, DD, HP);
    }

    // --- attention ---
    layernorm_bf16<<<TT, 256, 0, stream>>>(x, ln1_w, h1B, DD);

    dim3 gProj(DD / 64, TT / 128);
    gemm_bf16_wmma<<<gProj, 256, 0, stream>>>(h1B, wqT, qF, TT, DD, DD, nullptr, nullptr, 0, 0);
    gemm_bf16_wmma<<<gProj, 256, 0, stream>>>(h1B, wkT, kF, TT, DD, DD, nullptr, nullptr, 0, 0);
    gemm_bf16_wmma<<<gProj, 256, 0, stream>>>(h1B, wvT, vF, TT, DD, DD, nullptr, nullptr, 0, 0);

    rope_bf16<<<ceilDiv(TT * NHEAD * 32, 256), 256, 0, stream>>>(qF, kF, qB, kB);
    // Vt[d, t] = v[t, d]
    dim3 gVT(ceilDiv(TT, 256), DD);
    transpose_pad_cast<<<gVT, 256, 0, stream>>>(vF, vT, TT, DD, TT);

    flash_attn_wmma<<<NHEAD * (TT / 16), 32, 0, stream>>>(qB, kB, vT, oF);
    cast_f32_bf16<<<2048, 256, 0, stream>>>(oF, oB, TD);

    // d_out = x + o @ wo   (residual fused in epilogue)
    gemm_bf16_wmma<<<gProj, 256, 0, stream>>>(oB, woT, out, TT, DD, DD, x, nullptr, 0, 0);

    // --- MoE ---
    layernorm_bf16<<<TT, 256, 0, stream>>>(out, ln2_w, h2B, DD);
    gate_topk<<<TT, 32, 0, stream>>>(h2B, gatew, comb);

    dim3 gUp(HP / 64, TT / 128);     // [T,D] x Bt[HP,D] -> [T,HP]
    dim3 gDown(DD / 64, TT / 128);   // [T,HP] x Bt[D,HP] -> [T,D]
    for (int e = 0; e < EE; ++e) {
        const bf16* w1e = w1T + (size_t)e * HP * DD;
        const bf16* w3e = w3T + (size_t)e * HP * DD;
        const bf16* w2e = w2T + (size_t)e * DD * HP;
        gemm_bf16_wmma<<<gUp, 256, 0, stream>>>(h2B, w1e, c1, TT, HP, DD, nullptr, nullptr, 0, 0);
        gemm_bf16_wmma<<<gUp, 256, 0, stream>>>(h2B, w3e, c3, TT, HP, DD, nullptr, nullptr, 0, 0);
        silu_mul_bf16<<<4096, 256, 0, stream>>>(c1, c3, actB, THp);
        // d_out += combine[:,e] * (act @ w2_e)
        gemm_bf16_wmma<<<gDown, 256, 0, stream>>>(actB, w2e, out, TT, DD, HP,
                                                  nullptr, comb + e, EE, 1);
    }
}
